// VideoModelCoord_23244363006303
// MI455X (gfx1250) — matrix operand order repaired and checked by reading, not compiled, not run on hardware
//
#include <hip/hip_runtime.h>
#include <hip/hip_bf16.h>

typedef _Float16 v8h  __attribute__((ext_vector_type(8)));
typedef _Float16 v16h __attribute__((ext_vector_type(16)));
typedef float    v8f  __attribute__((ext_vector_type(8)));

// Problem constants: B=512, V=4, T=16, D=512, NC=174
// node rows = ((b*4+v)*16+t) -> 32768 ; edge rows = ((b*12+e)*16+t) -> 98304
// All GEMM M values are multiples of 128 -> no row bounds checks anywhere.

__constant__ int SENDc[12] = {0,0,0,1,1,1,2,2,2,3,3,3};
__constant__ int RECVc[12] = {1,2,3,0,2,3,0,1,3,0,1,2};

enum { F_BIAS = 1, F_RELU = 2, F_ELU = 4 };

// ---------------------------------------------------------------------------
// WMMA GEMM: C[M,N] = A[M,K](f16, lda) x Bw[K,N](f16, row-major)
// 128x128 block tile, 8 wave32s, each wave 64x32 via 4x2 v_wmma_f32_16x16x32_f16
// EDGE=false: requires K%32==0 and N%128==0 (branch-free staging/epilogue)
// EDGE=true : zero-fill bounds checks on K and N (used for K=4 and N=174)
// ---------------------------------------------------------------------------
template<bool EDGE>
__global__ __launch_bounds__(256) void k_gemm(
    const _Float16* __restrict__ A, const _Float16* __restrict__ Bw,
    const float* __restrict__ bias,
    float* __restrict__ outF, _Float16* __restrict__ outH,
    int M, int N, int K, int lda, int flags)
{
  __shared__ _Float16 sA[128][48];   // row-major A tile (48 halves = 96B rows)
  __shared__ _Float16 sB[128][48];   // transposed B tile: sB[n][k]

  const int m0   = blockIdx.x * 128;
  const int n0   = blockIdx.y * 128;
  const int tid  = threadIdx.x;
  const int lane = tid & 31;
  const int wv   = tid >> 5;
  const int wr   = wv >> 2;      // wave row 0..1  (64 rows each)
  const int wc   = wv & 3;       // wave col 0..3  (32 cols each)
  const int g    = lane >> 4;    // lane group 0/1
  const int lm   = lane & 15;

  const v8f vzero = {0.f,0.f,0.f,0.f,0.f,0.f,0.f,0.f};
  v8f acc[4][2];
  #pragma unroll
  for (int i = 0; i < 4; ++i)
    #pragma unroll
    for (int j = 0; j < 2; ++j) acc[i][j] = vzero;

  for (int k0 = 0; k0 < K; k0 += 32) {
    // prefetch next A K-slice into cache (global_prefetch_b8); uniform branch
    if (k0 + 32 < K)
      __builtin_prefetch((const void*)(A + (size_t)(m0 + (tid >> 1)) * lda + k0 + 32), 0, 0);

    if (!EDGE) {
      // ---- A: 128 rows x 32 halves; 2 unconditional b128 loads per thread
      #pragma unroll
      for (int it = 0; it < 2; ++it) {
        int c   = tid + it * 256;          // 0..511 chunks of 8 halves
        int row = c >> 2;
        int kk  = (c & 3) * 8;
        *(v8h*)&sA[row][kk] =
            *(const v8h*)(A + (size_t)(m0 + row) * lda + k0 + kk);
      }
      // ---- B: 32 k-rows x 128 n; vector global reads, transposed LDS stores
      #pragma unroll
      for (int it = 0; it < 2; ++it) {
        int idx = tid + it * 256;          // 0..511 chunks of 8 halves
        int k   = idx >> 4;                // 0..31
        int nc  = (idx & 15) * 8;          // 0..120
        v8h bv = *(const v8h*)(Bw + (size_t)(k0 + k) * N + n0 + nc);
        #pragma unroll
        for (int j = 0; j < 8; ++j) sB[nc + j][k] = bv[j];
      }
    } else {
      // ---- ragged K/N: scalar zero-fill staging (M never checked)
      #pragma unroll
      for (int it = 0; it < 2; ++it) {
        int c   = tid + it * 256;
        int row = c >> 2;
        int kk  = (c & 3) * 8;
        int gk  = k0 + kk;
        #pragma unroll
        for (int j = 0; j < 8; ++j)
          sA[row][kk + j] = (gk + j < K)
              ? A[(size_t)(m0 + row) * lda + gk + j] : (_Float16)0.f;
      }
      #pragma unroll
      for (int it = 0; it < 16; ++it) {
        int idx = tid + it * 256;
        int k = idx >> 7, n = idx & 127;
        int gk = k0 + k, gn = n0 + n;
        sB[n][k] = (gk < K && gn < N) ? Bw[(size_t)gk * N + gn] : (_Float16)0.f;
      }
    }
    __syncthreads();

    // ---- fragments: contiguous k order (element s = k 16g+s) in both A and B
    v16h af[4], bfv[2];
    #pragma unroll
    for (int mi = 0; mi < 4; ++mi) {
      const _Float16* pa = &sA[wr * 64 + mi * 16 + lm][0];
      v8h lo = *(const v8h*)(pa + 16 * g);       // K = 16g .. 16g+7
      v8h hi = *(const v8h*)(pa + 16 * g + 8);   // K = 16g+8 .. 16g+15
      af[mi] = __builtin_shufflevector(lo, hi,
                 0,1,2,3,4,5,6,7,8,9,10,11,12,13,14,15);
    }
    #pragma unroll
    for (int ni = 0; ni < 2; ++ni)
      bfv[ni] = *(const v16h*)&sB[wc * 32 + ni * 16 + lm][g * 16]; // K=16g..+15

    #pragma unroll
    for (int mi = 0; mi < 4; ++mi)
      #pragma unroll
      for (int ni = 0; ni < 2; ++ni)
        acc[mi][ni] = __builtin_amdgcn_wmma_f32_16x16x32_f16(
            false, af[mi], false, bfv[ni], (short)0, acc[mi][ni], false, false);
    __syncthreads();
  }

  // ---- epilogue (C layout: VGPR r, lane group g: row = r + 8g, col = lm)
  #pragma unroll
  for (int mi = 0; mi < 4; ++mi) {
    #pragma unroll
    for (int ni = 0; ni < 2; ++ni) {
      int col = n0 + wc * 32 + ni * 16 + lm;
      if (EDGE && col >= N) continue;
      float bv = (flags & F_BIAS) ? bias[col] : 0.f;
      #pragma unroll
      for (int r = 0; r < 8; ++r) {
        int row = m0 + wr * 64 + mi * 16 + g * 8 + r;
        float f = acc[mi][ni][r] + bv;
        if (flags & F_RELU) f = fmaxf(f, 0.f);
        if (flags & F_ELU)  f = (f > 0.f) ? f : (__expf(f) - 1.f);
        if (outH) outH[(size_t)row * N + col] = (_Float16)f;
        if (outF) outF[(size_t)row * N + col] = f;
      }
    }
  }
}

// ---------------------------------------------------------------------------
// BatchNorm batch statistics (biased var) per column; one thread per column.
// ---------------------------------------------------------------------------
__global__ void k_colstats(const float* __restrict__ y, int M, int N,
                           float* __restrict__ mean, float* __restrict__ rstd)
{
  int col = blockIdx.x * 256 + threadIdx.x;
  if (col >= N) return;
  float s = 0.f, sq = 0.f;
  for (int r = 0; r < M; ++r) {
    float v = y[(size_t)r * N + col];
    s += v; sq += v * v;
  }
  float m   = s / (float)M;
  float var = sq / (float)M - m * m;
  mean[col] = m;
  rstd[col] = rsqrtf(var + 1e-5f);
}

// BN apply (+optional ReLU), writes f16 into a possibly-wider destination
__global__ void k_bn_act(const float* __restrict__ y, int M, int N,
                         const float* __restrict__ gg, const float* __restrict__ bb,
                         const float* __restrict__ mean, const float* __restrict__ rstd,
                         _Float16* __restrict__ out, int ldo, int colofs, int act)
{
  int idx = blockIdx.x * 256 + threadIdx.x;
  if (idx >= M * N) return;
  int r = idx / N, c = idx - r * N;
  float v = (y[idx] - mean[c]) * rstd[c] * gg[c] + bb[c];
  if (act == 1) v = fmaxf(v, 0.f);
  out[(size_t)r * ldo + colofs + c] = (_Float16)v;
}

__global__ void k_cvt(const float* __restrict__ in, _Float16* __restrict__ out, int n)
{
  int i = blockIdx.x * 256 + threadIdx.x;
  if (i < n) out[i] = (_Float16)in[i];
}

// box_input [B,T,V,4] f32 -> x [((b*V+v)*T+t), 4] f16
__global__ void k_pack_x(const float* __restrict__ box, _Float16* __restrict__ out)
{
  int idx = blockIdx.x * 256 + threadIdx.x;     // 512*4*16*4 = 131072
  if (idx >= 131072) return;
  int c = idx & 3;
  int r = idx >> 2;
  int t = r & 15, v = (r >> 4) & 3, b = r >> 6;
  out[idx] = (_Float16)box[(((size_t)(b * 16 + t)) * 4 + v) * 4 + c];
}

// embedding (padding_idx=0) -> FUSIN cols [512,768)
__global__ void k_embed(const int* __restrict__ cats, const float* __restrict__ emb,
                        _Float16* __restrict__ fusin)
{
  int idx = blockIdx.x * 256 + threadIdx.x;     // 32768*256
  if (idx >= 32768 * 256) return;
  int c = idx & 255, r = idx >> 8;
  int t = r & 15, v = (r >> 4) & 3, b = r >> 6;
  int cat = cats[(b * 16 + t) * 4 + v];
  float val = (cat == 0) ? 0.f : emb[cat * 256 + c];
  fusin[(size_t)r * 768 + 512 + c] = (_Float16)val;
}

// node2edge: EIN[edge_row, 0:512]=bf[send], [512:1024]=bf[recv]
__global__ void k_node2edge(const _Float16* __restrict__ bf, _Float16* __restrict__ ein)
{
  long long idx = (long long)blockIdx.x * 256 + threadIdx.x;  // 98304*1024
  if (idx >= 100663296LL) return;
  int d = (int)(idx & 1023);
  long long r = idx >> 10;                   // ((b*12+e)*16+t)
  int t = (int)(r & 15);
  int e = (int)((r >> 4) % 12);
  int b = (int)(r / 192);
  int v = (d < 512) ? SENDc[e] : RECVc[e];
  int dd = d & 511;
  ein[idx] = bf[(((size_t)(b * 4 + v)) * 16 + t) * 512 + dd];
}

// edge2node: n[b,v,t,d] = mean over the 3 edges with recv==v
__global__ void k_edge2node(const _Float16* __restrict__ eo, _Float16* __restrict__ n0)
{
  int idx = blockIdx.x * 256 + threadIdx.x;   // 32768*512
  if (idx >= 16777216) return;
  int d = idx & 511;
  int r = idx >> 9;                           // (b*4+v)*16+t
  int t = r & 15, v = (r >> 4) & 3, b = r >> 6;
  float s = 0.f;
  #pragma unroll
  for (int sv = 0; sv < 4; ++sv) {
    if (sv == v) continue;
    int e = sv * 3 + (v > sv ? v - 1 : v);
    s += (float)eo[(((size_t)(b * 12 + e)) * 16 + t) * 512 + d];
  }
  n0[idx] = (_Float16)(s * (1.f / 3.f));
}

// copy bf into right half of t1 input
__global__ void k_copy_half(const _Float16* __restrict__ bf, _Float16* __restrict__ tin)
{
  int idx = blockIdx.x * 256 + threadIdx.x;   // 32768*512
  if (idx >= 16777216) return;
  int r = idx >> 9, c = idx & 511;
  tin[(size_t)r * 1024 + 512 + c] = bf[idx];
}

// mean over V=4 nodes -> feat f16 [512,512]
__global__ void k_mean(const _Float16* __restrict__ q, _Float16* __restrict__ feat)
{
  int idx = blockIdx.x * 256 + threadIdx.x;   // 512*512
  if (idx >= 262144) return;
  int d = idx & 511, b = idx >> 9;
  float s = 0.f;
  #pragma unroll
  for (int v = 0; v < 4; ++v) s += (float)q[((size_t)(b * 4 + v)) * 512 + d];
  feat[idx] = (_Float16)(s * 0.25f);
}

__global__ void k_labels(const int* __restrict__ lab, float* __restrict__ out)
{
  int i = blockIdx.x * 256 + threadIdx.x;
  if (i < 512) out[512 * 174 + i] = (float)lab[i];
}

// ---------------------------------------------------------------------------
extern "C" void kernel_launch(void* const* d_in, const int* in_sizes, int n_in,
                              void* d_out, int out_size, void* d_ws, size_t ws_size,
                              hipStream_t stream)
{
  (void)in_sizes; (void)n_in; (void)out_size; (void)ws_size;

  const int*   cats   = (const int*)  d_in[1];
  const float* box    = (const float*)d_in[2];
  const int*   label  = (const int*)  d_in[4];
  const float* emb    = (const float*)d_in[5];
  const float* c2f_w1 = (const float*)d_in[6];
  const float* c2f_g1 = (const float*)d_in[7];
  const float* c2f_b1 = (const float*)d_in[8];
  const float* c2f_w2 = (const float*)d_in[9];
  const float* c2f_g2 = (const float*)d_in[10];
  const float* c2f_b2 = (const float*)d_in[11];
  const float* fus_w  = (const float*)d_in[12];
  const float* fus_g  = (const float*)d_in[13];
  const float* fus_b  = (const float*)d_in[14];
  const float* m1_w1  = (const float*)d_in[15];
  const float* m1_b1  = (const float*)d_in[16];
  const float* m1_w2  = (const float*)d_in[17];
  const float* m1_b2  = (const float*)d_in[18];
  const float* m1_g   = (const float*)d_in[19];
  const float* m1_bb  = (const float*)d_in[20];
  const float* m2_w1  = (const float*)d_in[21];
  const float* m2_b1  = (const float*)d_in[22];
  const float* m2_w2  = (const float*)d_in[23];
  const float* m2_b2  = (const float*)d_in[24];
  const float* m2_g   = (const float*)d_in[25];
  const float* m2_bb  = (const float*)d_in[26];
  const float* t1_w1  = (const float*)d_in[27];
  const float* t1_b1  = (const float*)d_in[28];
  const float* t1_w2  = (const float*)d_in[29];
  const float* t1_b2  = (const float*)d_in[30];
  const float* t1_g   = (const float*)d_in[31];
  const float* t1_bb  = (const float*)d_in[32];
  const float* ps_w1  = (const float*)d_in[33];
  const float* ps_g1  = (const float*)d_in[34];
  const float* ps_b1  = (const float*)d_in[35];
  const float* ps_w2  = (const float*)d_in[36];
  const float* ps_g2  = (const float*)d_in[37];
  const float* ps_b2  = (const float*)d_in[38];
  const float* tb_w1  = (const float*)d_in[39];
  const float* tb_g1  = (const float*)d_in[40];
  const float* tb_b1  = (const float*)d_in[41];
  const float* tb_w2  = (const float*)d_in[42];
  const float* tb_g2  = (const float*)d_in[43];
  const float* tb_b2  = (const float*)d_in[44];
  const float* cl_w1  = (const float*)d_in[45];
  const float* cl_b1  = (const float*)d_in[46];
  const float* cl_w2  = (const float*)d_in[47];
  const float* cl_b2  = (const float*)d_in[48];
  const float* cl_w3  = (const float*)d_in[49];
  const float* cl_b3  = (const float*)d_in[50];

  // ---- workspace bump allocator
  char* base = (char*)d_ws;
  size_t cur = 0;
  auto alloc = [&](size_t bytes) -> void* {
    void* p = base + cur;
    cur = (cur + bytes + 255) & ~(size_t)255;
    return p;
  };

  const int Mn = 32768, Me = 98304, Mq = 2048, Mb = 512;

  _Float16* wc1  = (_Float16*)alloc((size_t)4 * 256 * 2);
  _Float16* wc2  = (_Float16*)alloc((size_t)256 * 512 * 2);
  _Float16* wfus = (_Float16*)alloc((size_t)768 * 512 * 2);
  _Float16* wm1a = (_Float16*)alloc((size_t)1024 * 512 * 2);
  _Float16* wm1b = (_Float16*)alloc((size_t)512 * 512 * 2);
  _Float16* wm2a = (_Float16*)alloc((size_t)512 * 512 * 2);
  _Float16* wm2b = (_Float16*)alloc((size_t)512 * 512 * 2);
  _Float16* wt1a = (_Float16*)alloc((size_t)1024 * 512 * 2);
  _Float16* wt1b = (_Float16*)alloc((size_t)512 * 512 * 2);
  _Float16* wps1 = (_Float16*)alloc((size_t)512 * 512 * 2);
  _Float16* wps2 = (_Float16*)alloc((size_t)512 * 512 * 2);
  _Float16* wtb1 = (_Float16*)alloc((size_t)8192 * 512 * 2);
  _Float16* wtb2 = (_Float16*)alloc((size_t)512 * 512 * 2);
  _Float16* wcl1 = (_Float16*)alloc((size_t)512 * 512 * 2);
  _Float16* wcl2 = (_Float16*)alloc((size_t)512 * 512 * 2);
  _Float16* wcl3 = (_Float16*)alloc((size_t)512 * 174 * 2);

  float* meanB = (float*)alloc(512 * 4);
  float* rstdB = (float*)alloc(512 * 4);

  _Float16* A0    = (_Float16*)alloc((size_t)Mn * 4 * 2);
  float*    Y     = (float*)   alloc((size_t)Me * 512 * 4);
  _Float16* G16   = (_Float16*)alloc((size_t)Me * 512 * 2);  // generic hidden buf
  _Float16* H1    = (_Float16*)alloc((size_t)Mn * 256 * 2);
  _Float16* FUSIN = (_Float16*)alloc((size_t)Mn * 768 * 2);
  _Float16* BF    = (_Float16*)alloc((size_t)Mn * 512 * 2);
  _Float16* EIN   = (_Float16*)alloc((size_t)Me * 1024 * 2);
  _Float16* EOUT  = (_Float16*)alloc((size_t)Me * 512 * 2);
  _Float16* N0    = (_Float16*)alloc((size_t)Mn * 512 * 2);
  _Float16* TIN   = (_Float16*)alloc((size_t)Mn * 1024 * 2);
  _Float16* P     = (_Float16*)alloc((size_t)Mn * 512 * 2);
  _Float16* P3    = (_Float16*)alloc((size_t)Mn * 512 * 2);  // == Q [2048,8192]
  _Float16* Q1    = (_Float16*)alloc((size_t)Mq * 512 * 2);
  _Float16* Q2    = (_Float16*)alloc((size_t)Mq * 512 * 2);
  _Float16* FEAT  = (_Float16*)alloc((size_t)Mb * 512 * 2);
  _Float16* C1    = (_Float16*)alloc((size_t)Mb * 512 * 2);
  _Float16* C2    = (_Float16*)alloc((size_t)Mb * 512 * 2);

  auto cvt = [&](const float* s, _Float16* dptr, int n) {
    k_cvt<<<(n + 255) / 256, 256, 0, stream>>>(s, dptr, n);
  };
  auto gemm = [&](const _Float16* Am, const _Float16* Wm, const float* bias,
                  float* oF, _Float16* oH, int M, int N, int K, int lda, int flags) {
    dim3 grid((M + 127) / 128, (N + 127) / 128);
    if ((K & 31) == 0 && (N & 127) == 0)
      k_gemm<false><<<grid, 256, 0, stream>>>(Am, Wm, bias, oF, oH, M, N, K, lda, flags);
    else
      k_gemm<true><<<grid, 256, 0, stream>>>(Am, Wm, bias, oF, oH, M, N, K, lda, flags);
  };
  auto stats_bn = [&](const float* Ym, int M, int N, const float* g, const float* b,
                      _Float16* out, int ldo, int colofs, int act) {
    k_colstats<<<(N + 255) / 256, 256, 0, stream>>>(Ym, M, N, meanB, rstdB);
    int tot = M * N;
    k_bn_act<<<(tot + 255) / 256, 256, 0, stream>>>(Ym, M, N, g, b, meanB, rstdB,
                                                    out, ldo, colofs, act);
  };

  // ---- convert weights to f16
  cvt(c2f_w1, wc1, 4 * 256);       cvt(c2f_w2, wc2, 256 * 512);
  cvt(fus_w,  wfus, 768 * 512);
  cvt(m1_w1,  wm1a, 1024 * 512);   cvt(m1_w2, wm1b, 512 * 512);
  cvt(m2_w1,  wm2a, 512 * 512);    cvt(m2_w2, wm2b, 512 * 512);
  cvt(t1_w1,  wt1a, 1024 * 512);   cvt(t1_w2, wt1b, 512 * 512);
  cvt(ps_w1,  wps1, 512 * 512);    cvt(ps_w2, wps2, 512 * 512);
  cvt(tb_w1,  wtb1, 8192 * 512);   cvt(tb_w2, wtb2, 512 * 512);
  cvt(cl_w1,  wcl1, 512 * 512);    cvt(cl_w2, wcl2, 512 * 512);
  cvt(cl_w3,  wcl3, 512 * 174);

  // ---- coord_to_feature: Linear-BN-ReLU x2
  k_pack_x<<<(131072 + 255) / 256, 256, 0, stream>>>(box, A0);
  gemm(A0, wc1, nullptr, Y, nullptr, Mn, 256, 4, 4, 0);       // EDGE (K=4)
  stats_bn(Y, Mn, 256, c2f_g1, c2f_b1, H1, 256, 0, 1);
  gemm(H1, wc2, nullptr, Y, nullptr, Mn, 512, 256, 256, 0);
  stats_bn(Y, Mn, 512, c2f_g2, c2f_b2, FUSIN, 768, 0, 1);     // left half of fusion in

  // ---- category embedding + fusion
  k_embed<<<(32768 * 256 + 255) / 256, 256, 0, stream>>>(cats, emb, FUSIN);
  gemm(FUSIN, wfus, nullptr, Y, nullptr, Mn, 512, 768, 768, 0);
  stats_bn(Y, Mn, 512, fus_g, fus_b, BF, 512, 0, 1);          // bf

  // ---- node2edge + MLP1 (Lin+ELU, Lin+ELU, BN)
  k_node2edge<<<(int)((100663296LL + 255) / 256), 256, 0, stream>>>(BF, EIN);
  gemm(EIN, wm1a, m1_b1, nullptr, G16, Me, 512, 1024, 1024, F_BIAS | F_ELU);
  gemm(G16, wm1b, m1_b2, Y, nullptr, Me, 512, 512, 512, F_BIAS | F_ELU);
  stats_bn(Y, Me, 512, m1_g, m1_bb, EOUT, 512, 0, 0);

  // ---- edge2node + MLP2
  k_edge2node<<<(16777216 + 255) / 256, 256, 0, stream>>>(EOUT, N0);
  gemm(N0, wm2a, m2_b1, nullptr, G16, Mn, 512, 512, 512, F_BIAS | F_ELU);
  gemm(G16, wm2b, m2_b2, Y, nullptr, Mn, 512, 512, 512, F_BIAS | F_ELU);
  stats_bn(Y, Mn, 512, m2_g, m2_bb, TIN, 1024, 0, 0);         // left half of t1 in

  // ---- temporal MLP on cat(node, bf)
  k_copy_half<<<(16777216 + 255) / 256, 256, 0, stream>>>(BF, TIN);
  gemm(TIN, wt1a, t1_b1, nullptr, G16, Mn, 512, 1024, 1024, F_BIAS | F_ELU);
  gemm(G16, wt1b, t1_b2, Y, nullptr, Mn, 512, 512, 512, F_BIAS | F_ELU);
  stats_bn(Y, Mn, 512, t1_g, t1_bb, P, 512, 0, 0);

  // ---- point_spatial_node_fusion: Linear-BN-ReLU x2
  gemm(P, wps1, nullptr, Y, nullptr, Mn, 512, 512, 512, 0);
  stats_bn(Y, Mn, 512, ps_g1, ps_b1, G16, 512, 0, 1);
  gemm(G16, wps2, nullptr, Y, nullptr, Mn, 512, 512, 512, 0);
  stats_bn(Y, Mn, 512, ps_g2, ps_b2, P3, 512, 0, 1);          // P3 == q [2048, 8192]

  // ---- t_box_feature_fusion over flattened time (reshape is free)
  gemm(P3, wtb1, nullptr, Y, nullptr, Mq, 512, 8192, 8192, 0);
  stats_bn(Y, Mq, 512, tb_g1, tb_b1, Q1, 512, 0, 1);
  gemm(Q1, wtb2, nullptr, Y, nullptr, Mq, 512, 512, 512, 0);
  stats_bn(Y, Mq, 512, tb_g2, tb_b2, Q2, 512, 0, 1);

  // ---- mean over boxes + classifier head
  k_mean<<<(262144 + 255) / 256, 256, 0, stream>>>(Q2, FEAT);
  gemm(FEAT, wcl1, cl_b1, nullptr, C1, Mb, 512, 512, 512, F_BIAS | F_RELU);
  gemm(C1, wcl2, cl_b2, nullptr, C2, Mb, 512, 512, 512, F_BIAS | F_RELU);
  gemm(C2, wcl3, cl_b3, (float*)d_out, nullptr, Mb, 174, 512, 512, F_BIAS); // EDGE (N=174)

  // ---- second tuple element: video_label
  k_labels<<<2, 256, 0, stream>>>(label, (float*)d_out);
}
